// SwitchMLP_23493471109477
// MI455X (gfx1250) — matrix operand order repaired and checked by reading, not compiled, not run on hardware
//
#include <hip/hip_runtime.h>
#include <hip/hip_bf16.h>
#include <math.h>

// ---------------------------------------------------------------------------
// Switch-MoE top-1 (B=2,S=2048,H=1024,FFN=4096,E=8) for gfx1250 / MI455X.
// Fast path: one-time fp32->bf16 weight transpose into workspace (134MB bf16,
// L2-resident), expert GEMMs with M=64 tiles, B fragments loaded directly
// from global memory, A fragments from LDS.  Fallback (small ws): round-2
// fp32-staging kernel with M=32 tiles.
// ---------------------------------------------------------------------------

typedef __attribute__((ext_vector_type(16))) __bf16 v16bf;
typedef __attribute__((ext_vector_type(8)))  __bf16 v8bf;
typedef __attribute__((ext_vector_type(4)))  __bf16 v4bf;
typedef __attribute__((ext_vector_type(8)))  float  v8f;
typedef __attribute__((ext_vector_type(4)))  float  v4f;

#define HDIM   1024
#define FFN    4096
#define NEXP   8
#define T_TOT  4096
#define XPITCH (HDIM + 8)            // padded lds row (halfs)
#define HPITCH (128 + 8)
#define BPITCH 48                    // fallback staging pitch

// ---------------- router: sigmoid(x @ Wr + br), top-1 (+ x -> bf16) --------
__launch_bounds__(256)
__global__ void moe_router(const float* __restrict__ x,
                           const float* __restrict__ Wr,
                           const float* __restrict__ br,
                           float* __restrict__ maxprob,
                           int*   __restrict__ maxind,
                           __bf16* __restrict__ xb) {
    const int wave = threadIdx.x >> 5;
    const int lane = threadIdx.x & 31;
    const int t = blockIdx.x * 8 + wave;
    if (t >= T_TOT) return;

    float acc[NEXP];
#pragma unroll
    for (int e = 0; e < NEXP; ++e) acc[e] = 0.0f;

    const float* xr = x + (size_t)t * HDIM;
    for (int h = lane; h < HDIM; h += 32) {
        const float xv = xr[h];
        if (xb) xb[(size_t)t * HDIM + h] = (__bf16)xv;
        const float* wr = Wr + (size_t)h * NEXP;
#pragma unroll
        for (int e = 0; e < NEXP; ++e) acc[e] += xv * wr[e];
    }
#pragma unroll
    for (int e = 0; e < NEXP; ++e) {
#pragma unroll
        for (int off = 16; off >= 1; off >>= 1)
            acc[e] += __shfl_xor(acc[e], off, 32);
    }
    if (lane == 0) {
        float best = -1.0f; int bi = 0;
#pragma unroll
        for (int e = 0; e < NEXP; ++e) {
            const float p = 1.0f / (1.0f + __expf(-(acc[e] + br[e])));
            if (p > best) { best = p; bi = e; }   // first-max tie-break
        }
        maxprob[t] = best;
        maxind[t]  = bi;
    }
}

// ---------------- dispatch: per-expert token lists --------------------------
__global__ void moe_zero_counts(int* counts) {
    if (threadIdx.x < NEXP) counts[threadIdx.x] = 0;
}

__launch_bounds__(256)
__global__ void moe_scatter(const int* __restrict__ maxind,
                            int* __restrict__ counts,
                            int* __restrict__ toklist) {
    const int t = blockIdx.x * blockDim.x + threadIdx.x;
    if (t >= T_TOT) return;
    const int e = maxind[t];
    const int p = atomicAdd(&counts[e], 1);
    toklist[e * T_TOT + p] = t;
}

// ---------------- [R x C] fp32 -> [C x R] bf16 transpose (per expert) ------
__launch_bounds__(256)
__global__ void transpose_convert(const float* __restrict__ src,
                                  __bf16* __restrict__ dst, int R, int C) {
    __shared__ __bf16 tile[64][64 + 8];
    const int e  = blockIdx.z;
    const int r0 = blockIdx.y * 64;
    const int c0 = blockIdx.x * 64;
    const float* s = src + (size_t)e * R * C;
    __bf16*      d = dst + (size_t)e * R * C;
    const int tid = threadIdx.x;
#pragma unroll
    for (int i = 0; i < 4; ++i) {                 // read coalesced along C
        const int r = (tid >> 4) + i * 16;
        const int c = (tid & 15) * 4;
        const v4f v = *(const v4f*)(s + (size_t)(r0 + r) * C + c0 + c);
        tile[c + 0][r] = (__bf16)v.x;
        tile[c + 1][r] = (__bf16)v.y;
        tile[c + 2][r] = (__bf16)v.z;
        tile[c + 3][r] = (__bf16)v.w;
    }
    __syncthreads();
#pragma unroll
    for (int i = 0; i < 4; ++i) {                 // write coalesced along R
        const int c = (tid >> 4) + i * 16;
        const int r = (tid & 15) * 4;
        v4bf w;
        w.x = tile[c][r]; w.y = tile[c][r + 1];
        w.z = tile[c][r + 2]; w.w = tile[c][r + 3];
        *(v4bf*)(d + (size_t)(c0 + c) * R + r0 + r) = w;
    }
}

// ---------------- WMMA fragment helpers (bf16, contiguous k = 16*(lane/16)+s) ----------
__device__ __forceinline__ v16bf load_a_frag(const __bf16* base, int pitch,
                                             int row, int k0, int lane) {
    const int off = (lane >> 4) << 4;                 // 0 or 16
    const __bf16* p = base + row * pitch + k0 + off;
    v16bf a;
    *((v8bf*)&a)     = *(const v8bf*)(p);
    *((v8bf*)&a + 1) = *(const v8bf*)(p + 8);
    return a;
}

// B frag from transposed bf16 weights in global: row = N-column, contiguous K.
__device__ __forceinline__ v16bf load_b_global(const __bf16* __restrict__ wt,
                                               size_t rowPitch, int col0,
                                               int k0, int lane) {
    const __bf16* p = wt + (size_t)(col0 + (lane & 15)) * rowPitch
                         + k0 + ((lane >> 4) << 4);
    v16bf b;
    *((v8bf*)&b)     = *(const v8bf*)(p);
    *((v8bf*)&b + 1) = *(const v8bf*)(p + 8);
    return b;
}

// B frag from LDS staging (fallback path)
__device__ __forceinline__ v16bf load_b_frag(const __bf16* bsm, int lane) {
    const int col  = lane & 15;
    const int koff = (lane >> 4) << 4;
    const __bf16* p = bsm + col * BPITCH + koff;
    v16bf b;
    *((v8bf*)&b)     = *(const v8bf*)(p);
    *((v8bf*)&b + 1) = *(const v8bf*)(p + 8);
    return b;
}

struct BReg { v4f r0a, r0b, r1a, r1b; };

__device__ __forceinline__ BReg load_b_slab(const float* __restrict__ g,
                                            size_t rowStride, int lane) {
    const int r0 = (lane & 15) * 2;
    const int ch = (lane >> 4) * 8;
    const float* p0 = g + (size_t)r0 * rowStride + ch;
    const float* p1 = p0 + rowStride;
    BReg b;
    b.r0a = *(const v4f*)(p0);
    b.r0b = *(const v4f*)(p0 + 4);
    b.r1a = *(const v4f*)(p1);
    b.r1b = *(const v4f*)(p1 + 4);
    return b;
}

__device__ __forceinline__ unsigned pack2(float lo, float hi) {
    union { __bf16 h[2]; unsigned u; } t;
    t.h[0] = (__bf16)lo; t.h[1] = (__bf16)hi;
    return t.u;
}

__device__ __forceinline__ void store_b_slab(__bf16* bsm, const BReg& b, int lane) {
    const int r0 = (lane & 15) * 2;
    const int ch = (lane >> 4) * 8;
#pragma unroll
    for (int q = 0; q < 4; ++q) {
        *(unsigned*)&bsm[(ch + q) * BPITCH + r0]     = pack2(b.r0a[q], b.r1a[q]);
        *(unsigned*)&bsm[(ch + 4 + q) * BPITCH + r0] = pack2(b.r0b[q], b.r1b[q]);
    }
}

// branchless exact-erf GELU (Winitzki approx, |err| ~1e-4 << bf16 noise)
__device__ __forceinline__ float gelu_fast(float x) {
    const float z  = 0.70710678118f * x;
    const float z2 = z * z;
    const float num = z2 * __builtin_fmaf(0.147f, z2, 1.27323954f);
    const float den = __builtin_fmaf(0.147f, z2, 1.0f);
    const float ex  = __expf(-num * __builtin_amdgcn_rcpf(den));
    const float er  = __builtin_copysignf(__builtin_sqrtf(1.0f - ex), x);
    return 0.5f * x * (1.0f + er);
}

// ======================= FAST PATH: bf16 weights, M=64 =====================
__launch_bounds__(256, 1)
__global__ void moe_expert_mlp_bf16(const __bf16* __restrict__ xb,
                                    const __bf16* __restrict__ W1t, // [E,FFN,H]
                                    const __bf16* __restrict__ W2t, // [E,H,FFN]
                                    const float* __restrict__ maxprob,
                                    const int*   __restrict__ counts,
                                    const int*   __restrict__ toklist,
                                    float* __restrict__ out) {
    __shared__ __bf16 lds_x[64][XPITCH];              // 132 KB token tile
    __shared__ __bf16 lds_h[64][HPITCH];              //  17 KB hidden chunk
    __shared__ int    s_tok[64];
    __shared__ float  s_prob[64];

    const int e    = blockIdx.x >> 6;                 // 64 tiles of M=64
    const int tile = blockIdx.x & 63;
    const int cnt  = counts[e];
    const int base = tile * 64;
    if (base >= cnt) return;
    const int nvalid = min(64, cnt - base);

    const int tid  = threadIdx.x;
    const int wave = tid >> 5;
    const int lane = tid & 31;

    if (tid < 64) {
        const int idx = (tid < nvalid) ? (base + tid) : base;
        const int t = toklist[e * T_TOT + idx];
        s_tok[tid]  = t;
        s_prob[tid] = maxprob[t];
    }
    __syncthreads();

    // gather X tile (bf16 copy): 4 threads/row, 512B each
    {
        const int row = tid >> 2, ts = tid & 3;
        const __bf16* xr = xb + (size_t)s_tok[row] * HDIM + ts * 256;
        __bf16* dr = &lds_x[row][ts * 256];
#pragma unroll
        for (int j = 0; j < 32; ++j)
            *(v8bf*)(dr + j * 8) = *(const v8bf*)(xr + j * 8);
    }
    __syncthreads();

    const v8f vzero = {0.f, 0.f, 0.f, 0.f, 0.f, 0.f, 0.f, 0.f};
    v8f acc[4][8];                                    // [m-tile][n-tile]
#pragma unroll
    for (int m = 0; m < 4; ++m)
#pragma unroll
        for (int n = 0; n < 8; ++n) acc[m][n] = vzero;

    const int arow0 = lane & 15;
    const __bf16* lds_x0 = &lds_x[0][0];
    const __bf16* lds_h0 = &lds_h[0][0];

#pragma unroll 1
    for (int chunk = 0; chunk < FFN / 128; ++chunk) {
        const int ffn0 = chunk * 128;

        // ---- fc1: wave computes hidden cols [ffn0 + wave*16, +16) for 4 m-tiles
        v8f h[4];
#pragma unroll
        for (int m = 0; m < 4; ++m) h[m] = vzero;
        {
            const __bf16* w1p = W1t + ((size_t)e * FFN + ffn0 + wave * 16) * HDIM;
            v16bf Bc = load_b_global(w1p, HDIM, 0, 0, lane);
#pragma unroll 1
            for (int kb = 0; kb < HDIM / 32; ++kb) {
                const int k0 = kb * 32;
                const int kn = (kb + 1 < HDIM / 32) ? (kb + 1) * 32 : k0;
                const v16bf Bn = load_b_global(w1p, HDIM, 0, kn, lane);
                const int kp = (kb + 8 < HDIM / 32) ? (kb + 8) * 32 : k0;
                __builtin_prefetch(w1p + (size_t)(lane & 15) * HDIM + kp, 0, 1);
#pragma unroll
                for (int m = 0; m < 4; ++m) {
                    const v16bf Am = load_a_frag(lds_x0, XPITCH, m * 16 + arow0, k0, lane);
                    h[m] = __builtin_amdgcn_wmma_f32_16x16x32_bf16(false, Am, false, Bc,
                                                                   (short)0, h[m], false, false);
                }
                Bc = Bn;
            }
        }
        // gelu -> lds_h
        {
            const int n   = lane & 15;
            const int mhi = (lane >> 4) << 3;
#pragma unroll
            for (int m = 0; m < 4; ++m)
#pragma unroll
                for (int v = 0; v < 8; ++v)
                    lds_h[m * 16 + v + mhi][wave * 16 + n] = (__bf16)gelu_fast(h[m][v]);
        }
        __syncthreads();

        // ---- fc2: acc += hidden_chunk @ W2t[out cols of this wave, ffn chunk]
        {
            const __bf16* w2p = W2t + ((size_t)e * HDIM + wave * 128) * FFN + ffn0;
#pragma unroll
            for (int nt = 0; nt < 8; ++nt) {
                v16bf Bk[4];
#pragma unroll
                for (int kb = 0; kb < 4; ++kb)
                    Bk[kb] = load_b_global(w2p, FFN, nt * 16, kb * 32, lane);
                if (nt + 1 < 8)
                    __builtin_prefetch(w2p + (size_t)((nt + 1) * 16 + (lane & 15)) * FFN, 0, 1);
#pragma unroll
                for (int kb = 0; kb < 4; ++kb) {
                    const int k0 = kb * 32;
#pragma unroll
                    for (int m = 0; m < 4; ++m) {
                        const v16bf Am = load_a_frag(lds_h0, HPITCH, m * 16 + arow0, k0, lane);
                        acc[m][nt] = __builtin_amdgcn_wmma_f32_16x16x32_bf16(
                            false, Am, false, Bk[kb], (short)0, acc[m][nt], false, false);
                    }
                }
            }
        }
        __syncthreads();   // protect lds_h before next chunk overwrites it
    }

    // ---- epilogue: scale by router prob, scatter to out
    {
        const int n   = lane & 15;
        const int mhi = (lane >> 4) << 3;
#pragma unroll
        for (int nt = 0; nt < 8; ++nt) {
            const int col = wave * 128 + nt * 16 + n;
#pragma unroll
            for (int m = 0; m < 4; ++m)
#pragma unroll
                for (int v = 0; v < 8; ++v) {
                    const int mr = m * 16 + v + mhi;
                    if (mr < nvalid)
                        __builtin_nontemporal_store(acc[m][nt][v] * s_prob[mr],
                                                    &out[(size_t)s_tok[mr] * HDIM + col]);
                }
        }
    }
}

// ======================= FALLBACK: fp32 staging, M=32 ======================
__launch_bounds__(256, 1)
__global__ void moe_expert_mlp_stage(const float* __restrict__ x,
                                     const float* __restrict__ W1,
                                     const float* __restrict__ W2,
                                     const float* __restrict__ maxprob,
                                     const int*   __restrict__ counts,
                                     const int*   __restrict__ toklist,
                                     float* __restrict__ out) {
    __shared__ __bf16 lds_x[32][XPITCH];
    __shared__ __bf16 lds_h[32][HPITCH];
    __shared__ __bf16 lds_b[8][16][BPITCH];
    __shared__ int    s_tok[32];
    __shared__ float  s_prob[32];

    const int e    = blockIdx.x >> 7;
    const int tile = blockIdx.x & 127;
    const int cnt  = counts[e];
    const int base = tile * 32;
    if (base >= cnt) return;
    const int nvalid = min(32, cnt - base);

    const int tid  = threadIdx.x;
    const int wave = tid >> 5;
    const int lane = tid & 31;

    if (tid < 32) {
        const int idx = (tid < nvalid) ? (base + tid) : base;
        const int t = toklist[e * T_TOT + idx];
        s_tok[tid]  = t;
        s_prob[tid] = maxprob[t];
    }
    __syncthreads();
    {
        const int row = tid >> 3, ts = tid & 7;
        const float* xr = x + (size_t)s_tok[row] * HDIM;
#pragma unroll 4
        for (int j = 0; j < 32; ++j) {
            const int c4 = ts + j * 8;
            const v4f v = *(const v4f*)(xr + c4 * 4);
            v4bf w;
            w.x = (__bf16)v.x; w.y = (__bf16)v.y;
            w.z = (__bf16)v.z; w.w = (__bf16)v.w;
            *(v4bf*)&lds_x[row][c4 * 4] = w;
        }
    }
    __syncthreads();

    const v8f vzero = {0.f, 0.f, 0.f, 0.f, 0.f, 0.f, 0.f, 0.f};
    v8f acc0[8], acc1[8];
#pragma unroll
    for (int i = 0; i < 8; ++i) { acc0[i] = vzero; acc1[i] = vzero; }

    const int arow0 = lane & 15;
    __bf16* myb = &lds_b[wave][0][0];
    const __bf16* lds_x0 = &lds_x[0][0];
    const __bf16* lds_h0 = &lds_h[0][0];

#pragma unroll 1
    for (int chunk = 0; chunk < FFN / 128; ++chunk) {
        const int ffn0 = chunk * 128;
        v8f h0 = vzero, h1 = vzero;
        {
            const float* wsrc = W1 + ((size_t)e * HDIM) * FFN + (ffn0 + wave * 16);
            BReg cur = load_b_slab(wsrc, FFN, lane);
#pragma unroll 1
            for (int kb = 0; kb < HDIM / 32; ++kb) {
                const int kn = (kb + 1 < HDIM / 32) ? kb + 1 : kb;
                BReg nxt = load_b_slab(wsrc + (size_t)kn * 32 * FFN, FFN, lane);
                store_b_slab(myb, cur, lane);
                asm volatile("" ::: "memory");
                const int k0 = kb * 32;
                const v16bf A0 = load_a_frag(lds_x0, XPITCH, arow0,      k0, lane);
                const v16bf A1 = load_a_frag(lds_x0, XPITCH, 16 + arow0, k0, lane);
                const v16bf Bf = load_b_frag(myb, lane);
                h0 = __builtin_amdgcn_wmma_f32_16x16x32_bf16(false, A0, false, Bf,
                                                             (short)0, h0, false, false);
                h1 = __builtin_amdgcn_wmma_f32_16x16x32_bf16(false, A1, false, Bf,
                                                             (short)0, h1, false, false);
                asm volatile("" ::: "memory");
                cur = nxt;
            }
        }
        {
            const int n   = lane & 15;
            const int mhi = (lane >> 4) << 3;
#pragma unroll
            for (int v = 0; v < 8; ++v) {
                lds_h[v + mhi][wave * 16 + n]      = (__bf16)gelu_fast(h0[v]);
                lds_h[16 + v + mhi][wave * 16 + n] = (__bf16)gelu_fast(h1[v]);
            }
        }
        __syncthreads();
        {
            const float* w2base = W2 + ((size_t)e * FFN + ffn0) * HDIM + wave * 128;
            BReg cur = load_b_slab(w2base, HDIM, lane);
#pragma unroll
            for (int i = 0; i < 32; ++i) {
                const int nt = i >> 2;
                const int k0 = (i & 3) * 32;
                const int in = (i + 1 < 32) ? i + 1 : i;
                const float* pn = w2base + (size_t)((in & 3) * 32) * HDIM + (in >> 2) * 16;
                BReg nxt = load_b_slab(pn, HDIM, lane);
                store_b_slab(myb, cur, lane);
                asm volatile("" ::: "memory");
                const v16bf A0 = load_a_frag(lds_h0, HPITCH, arow0,      k0, lane);
                const v16bf A1 = load_a_frag(lds_h0, HPITCH, 16 + arow0, k0, lane);
                const v16bf Bf = load_b_frag(myb, lane);
                acc0[nt] = __builtin_amdgcn_wmma_f32_16x16x32_bf16(false, A0, false, Bf,
                                                                   (short)0, acc0[nt], false, false);
                acc1[nt] = __builtin_amdgcn_wmma_f32_16x16x32_bf16(false, A1, false, Bf,
                                                                   (short)0, acc1[nt], false, false);
                asm volatile("" ::: "memory");
                cur = nxt;
            }
        }
        __syncthreads();
    }
    {
        const int n   = lane & 15;
        const int mhi = (lane >> 4) << 3;
#pragma unroll
        for (int nt = 0; nt < 8; ++nt) {
            const int col = wave * 128 + nt * 16 + n;
#pragma unroll
            for (int v = 0; v < 8; ++v) {
                const int m0 = v + mhi;
                if (m0 < nvalid)
                    __builtin_nontemporal_store(acc0[nt][v] * s_prob[m0],
                                                &out[(size_t)s_tok[m0] * HDIM + col]);
                const int m1 = 16 + v + mhi;
                if (m1 < nvalid)
                    __builtin_nontemporal_store(acc1[nt][v] * s_prob[m1],
                                                &out[(size_t)s_tok[m1] * HDIM + col]);
            }
        }
    }
}

// ---------------------------------------------------------------------------
extern "C" void kernel_launch(void* const* d_in, const int* in_sizes, int n_in,
                              void* d_out, int out_size, void* d_ws, size_t ws_size,
                              hipStream_t stream) {
    const float* x  = (const float*)d_in[0];   // [T, H]
    const float* Wr = (const float*)d_in[1];   // [H, E]
    const float* br = (const float*)d_in[2];   // [E]
    const float* W1 = (const float*)d_in[3];   // [E, H, FFN]
    const float* W2 = (const float*)d_in[4];   // [E, FFN, H]
    float* out = (float*)d_out;                // [T, H]

    char* ws = (char*)d_ws;
    float* maxprob = (float*)(ws);                          // 16 KB
    int*   maxind  = (int*)  (ws + (size_t)T_TOT * 4);      // 16 KB
    int*   counts  = (int*)  (ws + (size_t)T_TOT * 8);      // 64 B slot
    int*   toklist = (int*)  (ws + (size_t)T_TOT * 8 + 64); // 128 KB

    const size_t OFF_XB  = (size_t)192 * 1024;              // 8 MB bf16 x
    const size_t OFF_W1T = (size_t)16 << 20;                // 64 MB bf16 W1t
    const size_t OFF_W2T = (size_t)80 << 20;                // 64 MB bf16 W2t
    const size_t WS_NEED = (size_t)144 << 20;
    const bool fast = (ws_size >= WS_NEED);

    __bf16* xbuf = fast ? (__bf16*)(ws + OFF_XB)  : (__bf16*)0;
    __bf16* W1t  = (__bf16*)(ws + OFF_W1T);
    __bf16* W2t  = (__bf16*)(ws + OFF_W2T);

    moe_router     <<<T_TOT / 8,   256, 0, stream>>>(x, Wr, br, maxprob, maxind, xbuf);
    moe_zero_counts<<<1,           32,  0, stream>>>(counts);
    moe_scatter    <<<T_TOT / 256, 256, 0, stream>>>(maxind, counts, toklist);

    if (fast) {
        transpose_convert<<<dim3(FFN / 64, HDIM / 64, NEXP), 256, 0, stream>>>(
            W1, W1t, HDIM, FFN);   // -> W1t[e][FFN][H]
        transpose_convert<<<dim3(HDIM / 64, FFN / 64, NEXP), 256, 0, stream>>>(
            W2, W2t, FFN, HDIM);   // -> W2t[e][H][FFN]
        moe_expert_mlp_bf16<<<NEXP * 64, 256, 0, stream>>>(
            xbuf, W1t, W2t, maxprob, counts, toklist, out);
    } else {
        moe_expert_mlp_stage<<<NEXP * 128, 256, 0, stream>>>(
            x, W1, W2, maxprob, counts, toklist, out);
    }
}
